// R3MoERouter_18537078849839
// MI455X (gfx1250) — matrix operand order repaired and checked by reading, not compiled, not run on hardware
//
#include <hip/hip_runtime.h>
#include <hip/hip_bf16.h>

typedef __attribute__((ext_vector_type(16))) __bf16 v16bf;
typedef __attribute__((ext_vector_type(8)))  __bf16 v8bf;
typedef __attribute__((ext_vector_type(8)))  float  v8f;
typedef __attribute__((ext_vector_type(4)))  int    v4i;

#define NTOK 4096
#define DIM  1024
#define HDIM 2048
#define NEXP 8
#define TOPK 2
#define BM 128
#define BN 128
#define BK 32

union AFrag { v16bf v; v8bf h2[2]; };

typedef __attribute__((address_space(1))) v4i as1_v4i;
typedef __attribute__((address_space(3))) v4i as3_v4i;

// async copy of 16 bytes global->LDS (CDNA5 GLOBAL_LOAD_ASYNC_TO_LDS_B128),
// with safe fallback through VGPRs if the builtin is unavailable.
__device__ __forceinline__ void async_b128(const void* g, void* l) {
#if __has_builtin(__builtin_amdgcn_global_load_async_to_lds_b128)
    __builtin_amdgcn_global_load_async_to_lds_b128(
        (as1_v4i*)(unsigned long long)g,
        (as3_v4i*)(unsigned int)(unsigned long long)l,
        0, 0);
#else
    *(v8bf*)l = *(const v8bf*)g;
#endif
}

__device__ __forceinline__ void wait_async() {
#if __has_builtin(__builtin_amdgcn_global_load_async_to_lds_b128)
#if __has_builtin(__builtin_amdgcn_s_wait_asynccnt)
    __builtin_amdgcn_s_wait_asynccnt(0);
#else
    asm volatile("s_wait_asynccnt 0" ::: "memory");
#endif
#endif
}

// ---------------------------------------------------------------------------
// zero output accumulation region + ws counters
// ---------------------------------------------------------------------------
__global__ __launch_bounds__(256) void moe_zero_kernel(float* __restrict__ y, int n,
                                                       float* __restrict__ ctr, int nctr) {
    int i = blockIdx.x * 256 + threadIdx.x;
    if (i < n) y[i] = 0.0f;
    if (i < nctr) ctr[i] = 0.0f;
}

// ---------------------------------------------------------------------------
// f32 -> bf16 conversion (x, W1, W2 precompute); n multiple of 8
// ---------------------------------------------------------------------------
__global__ __launch_bounds__(256) void moe_cvt_kernel(const float* __restrict__ in,
                                                      __bf16* __restrict__ out, int n) {
    int i = (blockIdx.x * 256 + threadIdx.x) * 8;
    if (i + 8 <= n) {
        const float4* p = reinterpret_cast<const float4*>(in + i);
        float4 a = p[0], b = p[1];
        v8bf o;
        o[0]=(__bf16)a.x; o[1]=(__bf16)a.y; o[2]=(__bf16)a.z; o[3]=(__bf16)a.w;
        o[4]=(__bf16)b.x; o[5]=(__bf16)b.y; o[6]=(__bf16)b.z; o[7]=(__bf16)b.w;
        *(v8bf*)(out + i) = o;
    }
}

// ---------------------------------------------------------------------------
// router: scores = x @ router_w^T, top-2, gate, loss stats, expert token lists
// one wave32 per token, router_w cached in LDS (32 KB)
// ---------------------------------------------------------------------------
__global__ __launch_bounds__(256) void moe_router_kernel(
    const float* __restrict__ x, const float* __restrict__ rw,
    const float* __restrict__ train_scores,
    float* __restrict__ outScores, float* __restrict__ outTopk,
    float* __restrict__ outGate,
    float* __restrict__ fsum, float* __restrict__ psum,
    int* __restrict__ counts, int* __restrict__ tlist)
{
    __shared__ float rwl[NEXP * DIM];
    int tid = threadIdx.x;
    for (int i = tid; i < NEXP * DIM; i += 256) rwl[i] = rw[i];
    __syncthreads();

    int warp = tid >> 5, lane = tid & 31;
    int t = blockIdx.x * 8 + warp;

    float acc[NEXP];
#pragma unroll
    for (int e = 0; e < NEXP; ++e) acc[e] = 0.0f;
    const float* xp = x + (size_t)t * DIM;
    for (int d = lane; d < DIM; d += 32) {
        float xv = xp[d];
#pragma unroll
        for (int e = 0; e < NEXP; ++e) acc[e] += xv * rwl[e * DIM + d];
    }
#pragma unroll
    for (int off = 16; off > 0; off >>= 1) {
#pragma unroll
        for (int e = 0; e < NEXP; ++e) acc[e] += __shfl_down(acc[e], off, 32);
    }

    if (lane == 0) {
        float mx = acc[0];
#pragma unroll
        for (int e = 1; e < NEXP; ++e) mx = fmaxf(mx, acc[e]);
        float se = 0.0f, p[NEXP];
#pragma unroll
        for (int e = 0; e < NEXP; ++e) { p[e] = __expf(acc[e] - mx); se += p[e]; }
        int i1 = 0;
#pragma unroll
        for (int e = 1; e < NEXP; ++e) if (acc[e] > acc[i1]) i1 = e;
        int i2 = (i1 == 0) ? 1 : 0;
#pragma unroll
        for (int e = 0; e < NEXP; ++e) if (e != i1 && acc[e] > acc[i2]) i2 = e;
        float tm = train_scores[0];
#pragma unroll
        for (int e = 1; e < NEXP; ++e) tm = fmaxf(tm, train_scores[e]);
        float tw[NEXP], tsum = 0.0f;
#pragma unroll
        for (int e = 0; e < NEXP; ++e) { tw[e] = __expf(train_scores[e] - tm); tsum += tw[e]; }
        float gs  = (tw[i1] + tw[i2]) / tsum;
        float inv = 1.0f / fmaxf(gs, 1e-8f);
#pragma unroll
        for (int e = 0; e < NEXP; ++e) {
            outScores[(size_t)t * NEXP + e] = acc[e];
            float g = (e == i1 || e == i2) ? (tw[e] / tsum) * inv : 0.0f;
            outGate[(size_t)t * NEXP + e] = g;
            atomicAdd(&psum[e], p[e] / se);
        }
        outTopk[t * TOPK + 0] = (float)i1;
        outTopk[t * TOPK + 1] = (float)i2;
        atomicAdd(&fsum[i1], 1.0f);
        atomicAdd(&fsum[i2], 1.0f);
        int p1 = atomicAdd(&counts[i1], 1); tlist[i1 * NTOK + p1] = t;
        int p2 = atomicAdd(&counts[i2], 1); tlist[i2 * NTOK + p2] = t;
    }
}

// ---------------------------------------------------------------------------
// finalize: expert row offsets (prefix sum), router_loss, snapshot
// ---------------------------------------------------------------------------
__global__ void moe_finalize_kernel(const float* __restrict__ train_scores,
                                    const float* __restrict__ fsum,
                                    const float* __restrict__ psum,
                                    const int* __restrict__ counts,
                                    int* __restrict__ offsets,
                                    float* __restrict__ outLoss,
                                    float* __restrict__ outSnap)
{
    int tid = threadIdx.x;
    if (tid == 0) {
        int off = 0;
        float loss = 0.0f;
        for (int e = 0; e < NEXP; ++e) {
            offsets[e] = off;
            off += counts[e];
            loss += (fsum[e] / (float)NTOK) * (psum[e] / (float)NTOK);
        }
        *outLoss = (float)NEXP * loss;
    }
    if (tid < NEXP) outSnap[tid] = train_scores[tid];
}

// ---------------------------------------------------------------------------
// stage helpers: 16 elements/thread into LDS tile
// ---------------------------------------------------------------------------
__device__ __forceinline__ void stage_cvt16(const float* __restrict__ src,
                                            __bf16* __restrict__ dst, bool valid) {
    float f[16];
    if (valid) {
        const float4* p = reinterpret_cast<const float4*>(src);
#pragma unroll
        for (int q = 0; q < 4; ++q) {
            float4 u = p[q];
            f[q*4+0]=u.x; f[q*4+1]=u.y; f[q*4+2]=u.z; f[q*4+3]=u.w;
        }
    } else {
#pragma unroll
        for (int q = 0; q < 16; ++q) f[q] = 0.0f;
    }
#pragma unroll
    for (int q = 0; q < 16; ++q) dst[q] = (__bf16)f[q];
}

__device__ __forceinline__ void stage_copy16(const __bf16* __restrict__ src,
                                             __bf16* __restrict__ dst) {
    async_b128(src, dst);
    async_b128(src + 8, dst + 8);
}

// ---------------------------------------------------------------------------
// FFN1: h_compact[off_e + r] = gelu(x[tok] @ W1[e] + b1[e]) in bf16
// gathered grouped GEMM, v_wmma_f32_16x16x32_bf16
// 8 waves, BM=128 BN=128 BK=32; wave tile 32x64 -> 8 WMMA per k-step
// ---------------------------------------------------------------------------
template <bool PRE>
__global__ __launch_bounds__(256) void moe_ffn1_kernel(
    const float* __restrict__ xf, const __bf16* __restrict__ xb,
    const float* __restrict__ W1f, const __bf16* __restrict__ W1b,
    const float* __restrict__ b1,
    const int* __restrict__ counts, const int* __restrict__ offsets,
    const int* __restrict__ tlist, __bf16* __restrict__ hbuf)
{
    int e   = blockIdx.z;
    int cnt = counts[e];
    int m0  = blockIdx.y * BM;
    if (m0 >= cnt) return;
    int n0  = blockIdx.x * BN;

    __shared__ __bf16 At[BM * BK];
    __shared__ __bf16 Bt[BK * BN];
    __shared__ int    toks[BM];

    int tid = threadIdx.x;
    if (tid < BM) {
        int r = m0 + tid;
        toks[tid] = (r < cnt) ? tlist[e * NTOK + r] : -1;
    }
    __syncthreads();

    int wid = tid >> 5, lane = tid & 31;
    int wm = (wid & 3) * 32;
    int wn = (wid >> 2) * 64;
    int half = lane >> 4;

    v8f acc[2][4];
#pragma unroll
    for (int mi = 0; mi < 2; ++mi)
#pragma unroll
        for (int j = 0; j < 4; ++j)
#pragma unroll
            for (int r = 0; r < 8; ++r) acc[mi][j][r] = 0.0f;

    int arow = tid >> 1, acol = (tid & 1) * 16;  // A: 128x32, 16 elems/thread
    int brow = tid >> 3, bcol = (tid & 7) * 16;  // B: 32x128, 16 elems/thread
    int tok  = toks[arow];

    for (int k0 = 0; k0 < DIM; k0 += BK) {
        // stage A (gathered x rows)
        __bf16* ldsA = &At[arow * BK + acol];
        if constexpr (PRE) {
            if (tok >= 0)
                stage_copy16(xb + (size_t)tok * DIM + k0 + acol, ldsA);
        } else {
            stage_cvt16(xf + (size_t)(tok >= 0 ? tok : 0) * DIM + k0 + acol, ldsA, tok >= 0);
        }
        // stage B (W1 tile)
        __bf16* ldsB = &Bt[brow * BN + bcol];
        if constexpr (PRE) {
            stage_copy16(W1b + (size_t)e * DIM * HDIM + (size_t)(k0 + brow) * HDIM + n0 + bcol, ldsB);
        } else {
            const float* p = W1f + (size_t)e * DIM * HDIM + (size_t)(k0 + brow) * HDIM + n0 + bcol;
            if (k0 + BK < DIM) __builtin_prefetch(p + (size_t)BK * HDIM, 0, 1);
            stage_cvt16(p, ldsB, true);
        }
        if constexpr (PRE) wait_async();
        __syncthreads();

        AFrag afr[2];
#pragma unroll
        for (int mi = 0; mi < 2; ++mi) {
            int am = wm + mi * 16 + (lane & 15);
            afr[mi].h2[0] = *(const v8bf*)&At[am * BK + half * 8];
            afr[mi].h2[1] = *(const v8bf*)&At[am * BK + 16 + half * 8];
        }
#pragma unroll
        for (int j = 0; j < 4; ++j) {
            v16bf bfr;
#pragma unroll
            for (int s = 0; s < 16; ++s)
                bfr[s] = Bt[((s & 7) + 8 * half + 16 * (s >> 3)) * BN + wn + 16 * j + (lane & 15)];
#pragma unroll
            for (int mi = 0; mi < 2; ++mi)
                acc[mi][j] = __builtin_amdgcn_wmma_f32_16x16x32_bf16(
                    false, afr[mi].v, false, bfr, (short)0, acc[mi][j], false, false);
        }
        __syncthreads();
    }

    // epilogue: bias + exact gelu, store compact bf16 h
    const float* b1e = b1 + (size_t)e * HDIM;
    int hoff = offsets[e];
#pragma unroll
    for (int mi = 0; mi < 2; ++mi)
#pragma unroll
        for (int j = 0; j < 4; ++j) {
            int n = n0 + wn + 16 * j + (lane & 15);
            float bias = b1e[n];
#pragma unroll
            for (int r = 0; r < 8; ++r) {
                int gr = m0 + wm + mi * 16 + r + 8 * half;
                if (gr < cnt) {
                    float v = acc[mi][j][r] + bias;
                    float ge = 0.5f * v * (1.0f + erff(v * 0.70710678f));
                    hbuf[(size_t)(hoff + gr) * HDIM + n] = (__bf16)ge;
                }
            }
        }
}

// ---------------------------------------------------------------------------
// FFN2: out[tok] += gate[tok,e] * (h_compact @ W2[e] + b2[e])  (atomic scatter)
// ---------------------------------------------------------------------------
template <bool PRE>
__global__ __launch_bounds__(256) void moe_ffn2_kernel(
    const __bf16* __restrict__ hbuf,
    const float* __restrict__ W2f, const __bf16* __restrict__ W2b,
    const float* __restrict__ b2,
    const int* __restrict__ counts, const int* __restrict__ offsets,
    const int* __restrict__ tlist, const float* __restrict__ gate,
    float* __restrict__ outY)
{
    int e   = blockIdx.z;
    int cnt = counts[e];
    int m0  = blockIdx.y * BM;
    if (m0 >= cnt) return;
    int n0  = blockIdx.x * BN;

    __shared__ __bf16 At[BM * BK];
    __shared__ __bf16 Bt[BK * BN];
    __shared__ int    toks[BM];
    __shared__ float  gts[BM];

    int tid = threadIdx.x;
    if (tid < BM) {
        int r = m0 + tid;
        int t = (r < cnt) ? tlist[e * NTOK + r] : -1;
        toks[tid] = t;
        gts[tid]  = (t >= 0) ? gate[(size_t)t * NEXP + e] : 0.0f;
    }
    __syncthreads();

    int wid = tid >> 5, lane = tid & 31;
    int wm = (wid & 3) * 32;
    int wn = (wid >> 2) * 64;
    int half = lane >> 4;

    v8f acc[2][4];
#pragma unroll
    for (int mi = 0; mi < 2; ++mi)
#pragma unroll
        for (int j = 0; j < 4; ++j)
#pragma unroll
            for (int r = 0; r < 8; ++r) acc[mi][j][r] = 0.0f;

    int hoff = offsets[e];
    int arow = tid >> 1, acol = (tid & 1) * 16;
    int brow = tid >> 3, bcol = (tid & 7) * 16;
    bool avalid = (m0 + arow) < cnt;

    for (int k0 = 0; k0 < HDIM; k0 += BK) {
        // stage A (compact bf16 h rows) -- always async copy
        __bf16* ldsA = &At[arow * BK + acol];
        if (avalid)
            stage_copy16(hbuf + (size_t)(hoff + m0 + arow) * HDIM + k0 + acol, ldsA);
        // stage B (W2 tile)
        __bf16* ldsB = &Bt[brow * BN + bcol];
        if constexpr (PRE) {
            stage_copy16(W2b + (size_t)e * HDIM * DIM + (size_t)(k0 + brow) * DIM + n0 + bcol, ldsB);
        } else {
            const float* p = W2f + (size_t)e * HDIM * DIM + (size_t)(k0 + brow) * DIM + n0 + bcol;
            if (k0 + BK < HDIM) __builtin_prefetch(p + (size_t)BK * DIM, 0, 1);
            stage_cvt16(p, ldsB, true);
        }
        wait_async();
        __syncthreads();

        AFrag afr[2];
#pragma unroll
        for (int mi = 0; mi < 2; ++mi) {
            int am = wm + mi * 16 + (lane & 15);
            afr[mi].h2[0] = *(const v8bf*)&At[am * BK + half * 8];
            afr[mi].h2[1] = *(const v8bf*)&At[am * BK + 16 + half * 8];
        }
#pragma unroll
        for (int j = 0; j < 4; ++j) {
            v16bf bfr;
#pragma unroll
            for (int s = 0; s < 16; ++s)
                bfr[s] = Bt[((s & 7) + 8 * half + 16 * (s >> 3)) * BN + wn + 16 * j + (lane & 15)];
#pragma unroll
            for (int mi = 0; mi < 2; ++mi)
                acc[mi][j] = __builtin_amdgcn_wmma_f32_16x16x32_bf16(
                    false, afr[mi].v, false, bfr, (short)0, acc[mi][j], false, false);
        }
        __syncthreads();
    }

    // epilogue: gate-weighted atomic scatter into output
    const float* b2e = b2 + (size_t)e * DIM;
#pragma unroll
    for (int mi = 0; mi < 2; ++mi)
#pragma unroll
        for (int j = 0; j < 4; ++j) {
            int n = n0 + wn + 16 * j + (lane & 15);
            float bias = b2e[n];
#pragma unroll
            for (int r = 0; r < 8; ++r) {
                int ml = wm + mi * 16 + r + 8 * half;
                int gr = m0 + ml;
                if (gr < cnt) {
                    atomicAdd(&outY[(size_t)toks[ml] * DIM + n],
                              gts[ml] * (acc[mi][j][r] + bias));
                }
            }
        }
}

// ---------------------------------------------------------------------------
// host launch
// ---------------------------------------------------------------------------
extern "C" void kernel_launch(void* const* d_in, const int* in_sizes, int n_in,
                              void* d_out, int out_size, void* d_ws, size_t ws_size,
                              hipStream_t stream) {
    (void)in_sizes; (void)n_in; (void)out_size;

    const float* x  = (const float*)d_in[0];
    const float* rw = (const float*)d_in[1];
    const float* W1 = (const float*)d_in[2];
    const float* b1 = (const float*)d_in[3];
    const float* W2 = (const float*)d_in[4];
    const float* b2 = (const float*)d_in[5];
    const float* ts = (const float*)d_in[6];

    float* out       = (float*)d_out;
    float* outY      = out;
    float* outLoss   = outY + (size_t)NTOK * DIM;
    float* outScores = outLoss + 1;
    float* outTopk   = outScores + (size_t)NTOK * NEXP;
    float* outGate   = outTopk + (size_t)NTOK * TOPK;
    float* outSnap   = outGate + (size_t)NTOK * NEXP;

    char* ws = (char*)d_ws;
    const size_t hbytes = (size_t)NTOK * TOPK * HDIM * sizeof(__bf16);  // 32 MB compact h
    const size_t xbytes = (size_t)NTOK * DIM * sizeof(__bf16);          // 8 MB
    const size_t wbytes = (size_t)NEXP * DIM * HDIM * sizeof(__bf16);   // 32 MB each
    const size_t metabytes = 256 + (size_t)NEXP * NTOK * sizeof(int);

    const bool pre = ws_size >= (hbytes + xbytes + 2 * wbytes + metabytes);

    __bf16* hbuf = (__bf16*)ws;
    __bf16* xb   = (__bf16*)(ws + hbytes);
    __bf16* w1b  = (__bf16*)(ws + hbytes + xbytes);
    __bf16* w2b  = (__bf16*)(ws + hbytes + xbytes + wbytes);
    size_t meta_off = pre ? (hbytes + xbytes + 2 * wbytes) : hbytes;
    int*   counts  = (int*)(ws + meta_off);
    int*   offsets = counts + 8;
    float* fsum    = (float*)(ws + meta_off + 64);
    float* psum    = fsum + 8;
    int*   tlist   = (int*)(ws + meta_off + 256);

    int nzero = NTOK * DIM;
    moe_zero_kernel<<<(nzero + 255) / 256, 256, 0, stream>>>(outY, nzero, (float*)counts, 64);

    if (pre) {
        int nx = NTOK * DIM;
        int nw = NEXP * DIM * HDIM;
        moe_cvt_kernel<<<(nx / 8 + 255) / 256, 256, 0, stream>>>(x, xb, nx);
        moe_cvt_kernel<<<(nw / 8 + 255) / 256, 256, 0, stream>>>(W1, w1b, nw);
        moe_cvt_kernel<<<(nw / 8 + 255) / 256, 256, 0, stream>>>(W2, w2b, nw);
    }

    moe_router_kernel<<<NTOK / 8, 256, 0, stream>>>(x, rw, ts, outScores, outTopk, outGate,
                                                    fsum, psum, counts, tlist);

    moe_finalize_kernel<<<1, 32, 0, stream>>>(ts, fsum, psum, counts, offsets, outLoss, outSnap);

    dim3 g1(HDIM / BN, NTOK / BM, NEXP);
    dim3 g2(DIM / BN, NTOK / BM, NEXP);
    if (pre) {
        moe_ffn1_kernel<true><<<g1, 256, 0, stream>>>(x, xb, W1, w1b, b1,
                                                      counts, offsets, tlist, hbuf);
        moe_ffn2_kernel<true><<<g2, 256, 0, stream>>>(hbuf, W2, w2b, b2,
                                                      counts, offsets, tlist, outGate, outY);
    } else {
        moe_ffn1_kernel<false><<<g1, 256, 0, stream>>>(x, xb, W1, w1b, b1,
                                                       counts, offsets, tlist, hbuf);
        moe_ffn2_kernel<false><<<g2, 256, 0, stream>>>(hbuf, W2, w2b, b2,
                                                       counts, offsets, tlist, outGate, outY);
    }
}
